// SlotAttention_28183575396900
// MI455X (gfx1250) — matrix operand order repaired and checked by reading, not compiled, not run on hardware
//
#include <hip/hip_runtime.h>
#include <hip/hip_bf16.h>

// ---------------------------------------------------------------------------
// Slot Attention on gfx1250: all GEMMs via v_wmma_f32_16x16x32_bf16 (wave32),
// K/V projection streams feats strips through LDS with the Tensor Data Mover
// (tensor_load_to_lds + s_wait_tensorcnt double buffering).
// ---------------------------------------------------------------------------

typedef __attribute__((ext_vector_type(16))) __bf16 bf16x16;
typedef __attribute__((ext_vector_type(8)))  __bf16 bf16x8;
typedef __attribute__((ext_vector_type(8)))  float  floatx8;
typedef __attribute__((ext_vector_type(4)))  unsigned int uint32x4;
typedef __attribute__((ext_vector_type(8)))  int  int32x8;
typedef __attribute__((ext_vector_type(4)))  int  int32x4;

static constexpr int Bb = 64;    // batch
static constexpr int Nn = 4096;  // tokens
static constexpr int Dd = 256;   // feature dim
static constexpr int Ss = 8;     // slots
static constexpr int Mm = 1024;  // mlp hidden

__device__ __forceinline__ __bf16 f2bf(float f) {
    unsigned u = __float_as_uint(f);
    u += 0x7FFFu + ((u >> 16) & 1u);           // round-to-nearest-even
    unsigned short h = (unsigned short)(u >> 16);
    return __builtin_bit_cast(__bf16, h);
}

__device__ __forceinline__ floatx8 wmma_bf16(bf16x16 a, bf16x16 b, floatx8 c) {
    return __builtin_amdgcn_wmma_f32_16x16x32_bf16(false, a, false, b,
                                                   (short)0, c, false, false);
}

// A-operand tile (16x32 bf16): per-lane row = caller-resolved (lane%16),
// K chunks [kb..kb+7] and [kb+16..kb+23] with kb = k0 + (lane/16)*8.
__device__ __forceinline__ bf16x16 load_a16(const __bf16* row, int k0, int half) {
    const int kb = k0 + half * 8;
    bf16x8 lo = *(const bf16x8*)(row + kb);
    bf16x8 hi = *(const bf16x8*)(row + kb + 16);
    bf16x16 t;
#pragma unroll
    for (int i = 0; i < 8; ++i) { t[i] = lo[i]; t[i + 8] = hi[i]; }
    return t;
}

// Same A layout but converting from an fp32 row on the fly.
__device__ __forceinline__ bf16x16 load_a_f32(const float* row, int k0, int half) {
    const int kb = k0 + half * 8;
    bf16x16 t;
#pragma unroll
    for (int i = 0; i < 8; ++i) {
        t[i]     = f2bf(row[kb + i]);
        t[i + 8] = f2bf(row[kb + 16 + i]);
    }
    return t;
}

// B-operand tile (32x16 bf16): per-lane col = (lane%16), K chunks
// [kb..kb+7] and [kb+16..kb+23] with kb = k0 + (lane/16)*8 (same as A).
__device__ __forceinline__ bf16x16 load_b16(const __bf16* row, int k0, int half) {
    const int kb = k0 + half * 8;
    bf16x8 lo = *(const bf16x8*)(row + kb);
    bf16x8 hi = *(const bf16x8*)(row + kb + 16);
    bf16x16 t;
#pragma unroll
    for (int i = 0; i < 8; ++i) { t[i] = lo[i]; t[i + 8] = hi[i]; }
    return t;
}

// ---------------------------------------------------------------------------
// Elementwise helpers
// ---------------------------------------------------------------------------
__global__ void cast_bf16_kernel(const float* __restrict__ in,
                                 __bf16* __restrict__ out, int n) {
    int i = blockIdx.x * blockDim.x + threadIdx.x;
    if (i < n) out[i] = f2bf(in[i]);
}

__global__ void zero_kernel(float* p, int n) {
    int i = blockIdx.x * blockDim.x + threadIdx.x;
    if (i < n) p[i] = 0.0f;
}

__global__ void add_kernel(float* __restrict__ dst, const float* __restrict__ src, int n) {
    int i = blockIdx.x * blockDim.x + threadIdx.x;
    if (i < n) dst[i] += src[i];
}

__global__ void normupd_kernel(float* __restrict__ U, const float* __restrict__ denom) {
    int i = blockIdx.x * blockDim.x + threadIdx.x;
    if (i < Bb * Ss * Dd) U[i] /= denom[i >> 8];
}

__global__ void gru_kernel(const float* __restrict__ gi, const float* __restrict__ gh,
                           float* __restrict__ slots) {
    int i = blockIdx.x * blockDim.x + threadIdx.x;
    if (i >= Bb * Ss * Dd) return;
    int row = i >> 8, d = i & 255;
    const float* gir = gi + (size_t)row * 768;
    const float* ghr = gh + (size_t)row * 768;
    float r = 1.0f / (1.0f + __expf(-(gir[d] + ghr[d])));
    float z = 1.0f / (1.0f + __expf(-(gir[256 + d] + ghr[256 + d])));
    float n = tanhf(gir[512 + d] + r * ghr[512 + d]);
    float h = slots[i];
    slots[i] = (1.0f - z) * n + z * h;
}

// LayerNorm over rows of width 256; one wave per row; optional f32/bf16 out.
__global__ __launch_bounds__(256) void ln_kernel(const float* __restrict__ X,
                                                 const float* __restrict__ g,
                                                 const float* __restrict__ bta,
                                                 float* __restrict__ Yf,
                                                 __bf16* __restrict__ Yb, int R) {
    const int lane = threadIdx.x & 31;
    const int row  = blockIdx.x * 8 + (threadIdx.x >> 5);
    if (row >= R) return;
    const float* x = X + (size_t)row * Dd;
    float v[8], s = 0.0f, s2 = 0.0f;
#pragma unroll
    for (int j = 0; j < 8; ++j) {
        float t = x[lane + 32 * j];
        v[j] = t; s += t; s2 += t * t;
    }
#pragma unroll
    for (int off = 16; off; off >>= 1) {
        s  += __shfl_xor(s,  off, 32);
        s2 += __shfl_xor(s2, off, 32);
    }
    const float mu   = s * (1.0f / Dd);
    const float var  = s2 * (1.0f / Dd) - mu * mu;
    const float rstd = rsqrtf(var + 1e-5f);
#pragma unroll
    for (int j = 0; j < 8; ++j) {
        int c = lane + 32 * j;
        float y = (v[j] - mu) * rstd * g[c] + bta[c];
        if (Yf) Yf[(size_t)row * Dd + c] = y;
        if (Yb) Yb[(size_t)row * Dd + c] = f2bf(y);
    }
}

// ---------------------------------------------------------------------------
// K/V projection:   k[b][n][d]  = feats @ Wk^T   (computed as k^T = Wk @ feats^T)
//                   vT[b][d][n] = (feats @ Wv^T)^T
// Each wave processes 4 strips of 16 rows; feats strips are DMA'd into LDS by
// the Tensor Data Mover with double buffering (TENSORcnt pipelining).
// ---------------------------------------------------------------------------
static constexpr int STRIPS_PER_WAVE = 4;

__global__ __launch_bounds__(128) void kv_gemm_kernel(const __bf16* __restrict__ feats,
                                                      const __bf16* __restrict__ Wk,
                                                      const __bf16* __restrict__ Wv,
                                                      __bf16* __restrict__ kout,
                                                      __bf16* __restrict__ vT) {
    __shared__ __bf16 ftile[4][2][16 * Dd];            // 4 waves x 2 buffers x 8KB
    const int lane   = threadIdx.x & 31;
    const int wv     = threadIdx.x >> 5;
    const int l16    = lane & 15;
    const int half   = lane >> 4;
    const int strip0 = (blockIdx.x * 4 + wv) * STRIPS_PER_WAVE;

    // Issue one TDM 2D tile load: 16 rows x 256 bf16, row stride 256, into LDS.
    auto issue_tdm = [&](int buf, int strip) {
        unsigned lds_off = (unsigned)(size_t)(&ftile[wv][buf][0]);   // flat low32 = LDS offset
        unsigned long long ga =
            (unsigned long long)(size_t)(feats + (size_t)strip * 16 * Dd);
        uint32x4 g0;
        g0[0] = 1u;                                     // count=1 (valid), user mode
        g0[1] = lds_off;                                // D#.lds_addr (bytes)
        g0[2] = (unsigned)(ga & 0xFFFFFFFFu);           // global_addr[31:0]
        g0[3] = (unsigned)((ga >> 32) & 0x1FFFFFFu) | (2u << 30);  // addr[56:32] | type=2
        int32x8 g1;
        g1[0] = 0x00010000;         // workgroup_mask=0, data_size=1 (2-byte elements)
        g1[1] = (int)(256u << 16);  // tensor_dim0 = 256 (bits 79:48, low half)
        g1[2] = (int)(16u  << 16);  // tensor_dim1 = 16  (bits 111:80, low half)
        g1[3] = (int)(256u << 16);  // tile_dim0   = 256 (bits 127:112)
        g1[4] = 16;                 // tile_dim1   = 16  (bits 143:128)
        g1[5] = 256;                // tensor_dim0_stride = 256 (bits 207:160)
        g1[6] = 0;
        g1[7] = 0;
        int32x4 z4 = {0, 0, 0, 0};            // groups 2/3 unused (2D tensor)
        int32x8 z8 = {0, 0, 0, 0, 0, 0, 0, 0};
        __builtin_amdgcn_tensor_load_to_lds(g0, g1, z4, z4, z8, 0);
    };

    issue_tdm(0, strip0);
    for (int i = 0; i < STRIPS_PER_WAVE; ++i) {
        const int strip = strip0 + i;
        if (i + 1 < STRIPS_PER_WAVE) {
            // prior iteration's DS reads of the buffer we are about to refill
            asm volatile("s_wait_dscnt 0x0" ::: "memory");
            issue_tdm((i + 1) & 1, strip + 1);
            asm volatile("s_wait_tensorcnt 0x1" ::: "memory");   // strip i ready
        } else {
            asm volatile("s_wait_tensorcnt 0x0" ::: "memory");
        }
        const int row0 = strip * 16;                  // flat row in [0, B*N)
        const int lrow = row0 + l16;
        const int b    = row0 >> 12;
        const int n0   = row0 & 4095;
        const __bf16* frow = &ftile[wv][i & 1][l16 * Dd];

        // v = feats @ Wv^T  ->  vT[b][d][n]
        for (int nt = 0; nt < 16; ++nt) {
            const __bf16* wrow = Wv + (size_t)(nt * 16 + l16) * Dd;
            floatx8 acc = {};
#pragma unroll
            for (int kc = 0; kc < 8; ++kc)
                acc = wmma_bf16(load_a16(frow, kc * 32, half),
                                load_b16(wrow, kc * 32, half), acc);
            const int d = nt * 16 + l16;              // C lane column
            bf16x8 pk;
#pragma unroll
            for (int r = 0; r < 8; ++r) pk[r] = f2bf(acc[r]);
            *(bf16x8*)(vT + ((size_t)(b * Dd + d) * Nn) + n0 + half * 8) = pk;
        }
        // k^T = Wk @ feats^T  ->  k[b][n][d] row-major
        for (int mt = 0; mt < 16; ++mt) {
            const __bf16* arow = Wk + (size_t)(mt * 16 + l16) * Dd;
            floatx8 acc = {};
#pragma unroll
            for (int kc = 0; kc < 8; ++kc)
                acc = wmma_bf16(load_a16(arow, kc * 32, half),
                                load_b16(frow, kc * 32, half), acc);
            bf16x8 pk;
#pragma unroll
            for (int r = 0; r < 8; ++r) pk[r] = f2bf(acc[r]);
            *(bf16x8*)(kout + (size_t)lrow * Dd + mt * 16 + half * 8) = pk;
        }
    }
}

// ---------------------------------------------------------------------------
// q = LN(slots) @ Wq^T, written bf16 padded to 16 slot-rows (rows 8..15 = 0).
// One wave per batch.
// ---------------------------------------------------------------------------
__global__ __launch_bounds__(256) void qproj_kernel(const float* __restrict__ snorm,
                                                    const __bf16* __restrict__ Wq,
                                                    __bf16* __restrict__ qb) {
    const int lane = threadIdx.x & 31;
    const int b    = blockIdx.x * 8 + (threadIdx.x >> 5);
    if (b >= Bb) return;
    const int l16 = lane & 15, half = lane >> 4;
    const float* arow = snorm + ((size_t)b * Ss + (l16 & 7)) * Dd;
    const bool realrow = (l16 < 8);
    for (int nt = 0; nt < 16; ++nt) {
        const __bf16* wrow = Wq + (size_t)(nt * 16 + l16) * Dd;
        floatx8 acc = {};
        for (int kc = 0; kc < 8; ++kc) {
            bf16x16 a;
            if (realrow) {
                a = load_a_f32(arow, kc * 32, half);
            } else {
#pragma unroll
                for (int i = 0; i < 16; ++i) a[i] = f2bf(0.0f);
            }
            acc = wmma_bf16(a, load_b16(wrow, kc * 32, half), acc);
        }
#pragma unroll
        for (int r = 0; r < 8; ++r) {
            int srow = half * 8 + r;     // padded rows come out as exact zeros
            qb[((size_t)b * 16 + srow) * Dd + nt * 16 + l16] = f2bf(acc[r]);
        }
    }
}

// ---------------------------------------------------------------------------
// Fused: logits = (q k^T)/sqrt(D), softmax over slots (lane-local!),
// attn += eps (written to d_out), partial denom + unnormalized updates
// (attn @ v) accumulated with f32 atomics.  grid = (B, 4), 8 waves/block,
// each wave owns 128 columns of N.
// ---------------------------------------------------------------------------
__global__ __launch_bounds__(256) void attn_updates_kernel(const __bf16* __restrict__ qb,
                                                           const __bf16* __restrict__ kmat,
                                                           const __bf16* __restrict__ vT,
                                                           float* __restrict__ U,
                                                           float* __restrict__ denom,
                                                           float* __restrict__ attn_out) {
    __shared__ __bf16 lds[8][16][32];
    const int lane   = threadIdx.x & 31;
    const int wv     = threadIdx.x >> 5;
    const int b      = blockIdx.x;
    const int n_base = blockIdx.y * 1024 + wv * 128;
    const int l16    = lane & 15;
    const int half   = lane >> 4;

    // cache q A-tiles (K = 256 -> 8 chunks)
    const __bf16* qrow = qb + ((size_t)b * 16 + l16) * Dd;
    bf16x16 aT[8];
#pragma unroll
    for (int kc = 0; kc < 8; ++kc) aT[kc] = load_a16(qrow, kc * 32, half);

    floatx8 upd[16];
#pragma unroll
    for (int i = 0; i < 16; ++i) upd[i] = (floatx8){};
    floatx8 dsum = {};

    for (int kc2 = 0; kc2 < 4; ++kc2) {                 // 32-column K-chunks
#pragma unroll
        for (int t = 0; t < 2; ++t) {
            const int n0 = n_base + kc2 * 32 + t * 16;
            const __bf16* krow = kmat + ((size_t)b * Nn + n0 + l16) * Dd;
            __builtin_prefetch(krow + 16 * Dd, 0, 1);   // global_prefetch_b8
            floatx8 acc = {};
#pragma unroll
            for (int kq = 0; kq < 8; ++kq)
                acc = wmma_bf16(aT[kq], load_b16(krow, kq * 32, half), acc);
            // lanes 0..15 hold column n with slot rows 0..7 in registers:
            // softmax over slots is lane-local.
            if (lane < 16) {
                float mx = -1e30f;
#pragma unroll
                for (int r = 0; r < 8; ++r) { acc[r] *= 0.0625f; mx = fmaxf(mx, acc[r]); }
                float s = 0.0f, p[8];
#pragma unroll
                for (int r = 0; r < 8; ++r) { p[r] = __expf(acc[r] - mx); s += p[r]; }
                const float inv = 1.0f / s;
                const int n = n0 + lane;
#pragma unroll
                for (int r = 0; r < 8; ++r) {
                    float a = p[r] * inv + 1e-8f;
                    dsum[r] += a;
                    attn_out[((size_t)b * Ss + r) * Nn + n] = a;
                    lds[wv][r][t * 16 + lane] = f2bf(a);
                }
            } else {
#pragma unroll
                for (int r = 0; r < 8; ++r) lds[wv][8 + r][t * 16 + l16] = f2bf(0.0f);
            }
        }
        // same-wave DS ordering before reading attn back in A layout
        asm volatile("s_wait_dscnt 0x0" ::: "memory");
        bf16x16 at = load_a16(&lds[wv][l16][0], 0, half);
        const int nk = n_base + kc2 * 32;
#pragma unroll
        for (int dt = 0; dt < 16; ++dt) {
            const __bf16* vrow = vT + ((size_t)(b * Dd + dt * 16 + l16)) * Nn + nk;
            upd[dt] = wmma_bf16(at, load_b16(vrow, 0, half), upd[dt]);
        }
        asm volatile("s_wait_dscnt 0x0" ::: "memory");
    }

    if (lane < 16) {   // only slot rows 0..7 are real
#pragma unroll
        for (int dt = 0; dt < 16; ++dt) {
            const int d = dt * 16 + lane;
#pragma unroll
            for (int r = 0; r < 8; ++r)
                atomicAdd(&U[((size_t)b * Ss + r) * Dd + d], upd[dt][r]);
        }
#pragma unroll
        for (int r = 0; r < 8; ++r) atomicAdd(&denom[b * Ss + r], dsum[r]);
    }
}

// ---------------------------------------------------------------------------
// Generic  C[M,Nout] = relu?(A[M,K](f32) * W[Nout,K](bf16)^T + bias)
// One wave per 16-row M-tile, looping over N-tiles.
// ---------------------------------------------------------------------------
__global__ __launch_bounds__(256) void gemm_kernel(const float* __restrict__ A,
                                                   const __bf16* __restrict__ W,
                                                   const float* __restrict__ bias,
                                                   float* __restrict__ C,
                                                   int M, int K, int Nout, int relu) {
    const int lane  = threadIdx.x & 31;
    const int mtile = blockIdx.x * 8 + (threadIdx.x >> 5);
    if (mtile * 16 >= M) return;
    const int l16 = lane & 15, half = lane >> 4;
    const float* arow = A + ((size_t)mtile * 16 + l16) * K;
    const int ntiles = Nout >> 4, kcs = K >> 5;
    for (int nt = 0; nt < ntiles; ++nt) {
        const __bf16* wrow = W + (size_t)(nt * 16 + l16) * K;
        floatx8 acc = {};
        for (int kc = 0; kc < kcs; ++kc)
            acc = wmma_bf16(load_a_f32(arow, kc * 32, half),
                            load_b16(wrow, kc * 32, half), acc);
        const int col = nt * 16 + l16;
        const float bv = bias ? bias[col] : 0.0f;
#pragma unroll
        for (int r = 0; r < 8; ++r) {
            float v = acc[r] + bv;
            if (relu) v = fmaxf(v, 0.0f);
            C[((size_t)mtile * 16 + half * 8 + r) * Nout + col] = v;
        }
    }
}

// ---------------------------------------------------------------------------
// Host-side orchestration
// ---------------------------------------------------------------------------
extern "C" void kernel_launch(void* const* d_in, const int* in_sizes, int n_in,
                              void* d_out, int out_size, void* d_ws, size_t ws_size,
                              hipStream_t stream) {
    const float* features   = (const float*)d_in[0];
    const float* slots_init = (const float*)d_in[1];
    const float* g_feat = (const float*)d_in[2];
    const float* b_feat = (const float*)d_in[3];
    const float* g_slot = (const float*)d_in[4];
    const float* b_slot = (const float*)d_in[5];
    const float* g_mlp  = (const float*)d_in[6];
    const float* b_mlp  = (const float*)d_in[7];
    const float* Wk = (const float*)d_in[8];
    const float* Wv = (const float*)d_in[9];
    const float* Wq = (const float*)d_in[10];
    const float* w_ih = (const float*)d_in[11];
    const float* w_hh = (const float*)d_in[12];
    const float* b_ih = (const float*)d_in[13];
    const float* b_hh = (const float*)d_in[14];
    const float* W1 = (const float*)d_in[15];
    const float* b1 = (const float*)d_in[16];
    const float* W2 = (const float*)d_in[17];
    const float* b2 = (const float*)d_in[18];
    const int num_iter = 3;  // reference: num_iter = 3

    char* ws = (char*)d_ws;
    size_t off = 0;
    auto alloc = [&](size_t bytes) { char* p = ws + off; off += (bytes + 255) & ~size_t(255); return p; };

    __bf16* feats_bf = (__bf16*)alloc((size_t)Bb * Nn * Dd * 2);
    __bf16* k_bf     = (__bf16*)alloc((size_t)Bb * Nn * Dd * 2);
    __bf16* vT_bf    = (__bf16*)alloc((size_t)Bb * Nn * Dd * 2);
    __bf16* Wk_bf    = (__bf16*)alloc((size_t)Dd * Dd * 2);
    __bf16* Wv_bf    = (__bf16*)alloc((size_t)Dd * Dd * 2);
    __bf16* Wq_bf    = (__bf16*)alloc((size_t)Dd * Dd * 2);
    __bf16* wih_bf   = (__bf16*)alloc((size_t)3 * Dd * Dd * 2);
    __bf16* whh_bf   = (__bf16*)alloc((size_t)3 * Dd * Dd * 2);
    __bf16* W1_bf    = (__bf16*)alloc((size_t)Mm * Dd * 2);
    __bf16* W2_bf    = (__bf16*)alloc((size_t)Dd * Mm * 2);
    float*  snorm    = (float*)alloc((size_t)Bb * Ss * Dd * 4);
    __bf16* q_bf     = (__bf16*)alloc((size_t)Bb * 16 * Dd * 2);
    float*  U        = (float*)alloc((size_t)Bb * Ss * Dd * 4);
    float*  denom    = (float*)alloc((size_t)Bb * Ss * 4);
    float*  slots    = (float*)alloc((size_t)Bb * Ss * Dd * 4);
    float*  gi       = (float*)alloc((size_t)Bb * Ss * 3 * Dd * 4);
    float*  gh       = (float*)alloc((size_t)Bb * Ss * 3 * Dd * 4);
    float*  h1       = (float*)alloc((size_t)Bb * Ss * Mm * 4);
    float*  m2       = (float*)alloc((size_t)Bb * Ss * Dd * 4);

    float* out_slots = (float*)d_out;                    // [B,S,D]
    float* out_attn  = (float*)d_out + Bb * Ss * Dd;     // [B,S,N]

    auto cast = [&](const float* src, __bf16* dst, int n) {
        cast_bf16_kernel<<<(n + 255) / 256, 256, 0, stream>>>(src, dst, n);
    };
    cast(Wk, Wk_bf, Dd * Dd);
    cast(Wv, Wv_bf, Dd * Dd);
    cast(Wq, Wq_bf, Dd * Dd);
    cast(w_ih, wih_bf, 3 * Dd * Dd);
    cast(w_hh, whh_bf, 3 * Dd * Dd);
    cast(W1, W1_bf, Mm * Dd);
    cast(W2, W2_bf, Dd * Mm);

    // LN(features) -> bf16
    ln_kernel<<<(Bb * Nn) / 8, 256, 0, stream>>>(features, g_feat, b_feat,
                                                 nullptr, feats_bf, Bb * Nn);
    // k, vT via TDM-pipelined WMMA GEMM: 16384 strips / (4 per wave) / (4 waves)
    kv_gemm_kernel<<<(Bb * Nn / 16) / (STRIPS_PER_WAVE * 4), 128, 0, stream>>>(
        feats_bf, Wk_bf, Wv_bf, k_bf, vT_bf);
    // slots <- slots_init
    (void)hipMemcpyAsync(slots, slots_init, (size_t)Bb * Ss * Dd * 4,
                         hipMemcpyDeviceToDevice, stream);

    const int nSD = Bb * Ss * Dd;   // 131072
    for (int it = 0; it < num_iter; ++it) {
        ln_kernel<<<(Bb * Ss) / 8, 256, 0, stream>>>(slots, g_slot, b_slot,
                                                     snorm, nullptr, Bb * Ss);
        qproj_kernel<<<Bb / 8, 256, 0, stream>>>(snorm, Wq_bf, q_bf);
        zero_kernel<<<(nSD + 255) / 256, 256, 0, stream>>>(U, nSD);
        zero_kernel<<<(Bb * Ss + 255) / 256, 256, 0, stream>>>(denom, Bb * Ss);
        attn_updates_kernel<<<dim3(Bb, 4), 256, 0, stream>>>(q_bf, k_bf, vT_bf,
                                                             U, denom, out_attn);
        normupd_kernel<<<(nSD + 255) / 256, 256, 0, stream>>>(U, denom);
        // GRU: gi = updates @ w_ih^T + b_ih ; gh = slots @ w_hh^T + b_hh
        gemm_kernel<<<(Bb * Ss / 16) / 8, 256, 0, stream>>>(U, wih_bf, b_ih, gi,
                                                            Bb * Ss, Dd, 3 * Dd, 0);
        gemm_kernel<<<(Bb * Ss / 16) / 8, 256, 0, stream>>>(slots, whh_bf, b_hh, gh,
                                                            Bb * Ss, Dd, 3 * Dd, 0);
        gru_kernel<<<(nSD + 255) / 256, 256, 0, stream>>>(gi, gh, slots);
        // MLP residual
        ln_kernel<<<(Bb * Ss) / 8, 256, 0, stream>>>(slots, g_mlp, b_mlp,
                                                     snorm, nullptr, Bb * Ss);
        gemm_kernel<<<(Bb * Ss / 16) / 8, 256, 0, stream>>>(snorm, W1_bf, b1, h1,
                                                            Bb * Ss, Dd, Mm, 1);
        gemm_kernel<<<(Bb * Ss / 16) / 8, 256, 0, stream>>>(h1, W2_bf, b2, m2,
                                                            Bb * Ss, Mm, Dd, 0);
        add_kernel<<<(nSD + 255) / 256, 256, 0, stream>>>(slots, m2, nSD);
    }
    (void)hipMemcpyAsync(out_slots, slots, (size_t)nSD * 4,
                         hipMemcpyDeviceToDevice, stream);
    (void)in_sizes; (void)n_in; (void)out_size; (void)ws_size;
}
